// cnx_classify_idcard_4_21723944583734
// MI455X (gfx1250) — matrix operand order repaired and checked by reading, not compiled, not run on hardware
//
#include <hip/hip_runtime.h>

// ---------------------------------------------------------------------------
// MobileNetV1 backbone + 256 per-class heads for MI455X (gfx1250, wave32).
// Dominant cost: per-head 512->1024 pointwise GEMM (105 GFLOP) -> bf16 WMMA.
// B-matrix strip staged per-workgroup into LDS via global_load_async_to_lds
// (ASYNCcnt path), A streamed from global with WGP-scope prefetch,
// BN+ReLU+avgpool fused in the GEMM epilogue.
// ---------------------------------------------------------------------------

#define BN_EPS 1e-5f

typedef __attribute__((ext_vector_type(16))) __bf16       v16bf;
typedef __attribute__((ext_vector_type(8)))  float        v8f;
typedef __attribute__((ext_vector_type(4)))  unsigned int u32x4;
typedef __attribute__((ext_vector_type(4)))  float        f32x4;

union FragBF { v16bf v; u32x4 q[2]; };
union AccF   { v8f  v; float f[8];  };

__device__ __forceinline__ unsigned short f2bf(float f) {
  unsigned u = __float_as_uint(f);
  u += 0x7fffu + ((u >> 16) & 1u);          // round-to-nearest-even
  return (unsigned short)(u >> 16);
}

// Async global->LDS copy of 16 bytes (CDNA5 GLOBAL_LOAD_ASYNC_TO_LDS_B128,
// tracked by ASYNCcnt). lds_off is the LDS byte address (low 32 bits of the
// generic pointer to a __shared__ object), gaddr a 64-bit global address.
__device__ __forceinline__ void async_copy_b128(unsigned lds_off,
                                                unsigned long long gaddr) {
  asm volatile("global_load_async_to_lds_b128 %0, %1, off"
               :: "v"(lds_off), "v"(gaddr)
               : "memory");
}

__device__ __forceinline__ void wait_asynccnt0() {
#if __has_builtin(__builtin_amdgcn_s_wait_asynccnt)
  __builtin_amdgcn_s_wait_asynccnt(0);
#else
  asm volatile("s_wait_asynccnt 0x0" ::: "memory");
#endif
}

// ------------------------- prep kernels ------------------------------------

// Fold eval-mode BN into (scale, shift) pairs: y = x*s + t
__global__ void k_bn_fold(const float* __restrict__ g, const float* __restrict__ b,
                          const float* __restrict__ m, const float* __restrict__ v,
                          float* __restrict__ st, int n) {
  int i = blockIdx.x * blockDim.x + threadIdx.x;
  if (i >= n) return;
  float s = g[i] * rsqrtf(v[i] + BN_EPS);
  st[2 * i]     = s;
  st[2 * i + 1] = b[i] - m[i] * s;
}

__global__ void k_zero(float* __restrict__ p, long n) {
  long i = blockIdx.x * (long)blockDim.x + threadIdx.x;
  if (i < n) p[i] = 0.0f;
}

// fp32 -> bf16 cast, 4 elements / thread (vectorized 16B read, 8B write)
__global__ void k_cast_bf16x4(const float* __restrict__ src,
                              unsigned short* __restrict__ dst, long n4) {
  long i = blockIdx.x * (long)blockDim.x + threadIdx.x;
  if (i >= n4) return;
  f32x4 x = *(const f32x4*)(src + 4 * i);
  unsigned long long pk =
      (unsigned long long)f2bf(x[0])        |
      ((unsigned long long)f2bf(x[1]) << 16) |
      ((unsigned long long)f2bf(x[2]) << 32) |
      ((unsigned long long)f2bf(x[3]) << 48);
  *(unsigned long long*)(dst + 4 * i) = pk;
}

// ------------------------- backbone (fp32, ~4% of FLOPs) -------------------

// stem: conv3x3 s2 pad1, 3->32, BN+ReLU. out [8,32,112,112]
__global__ __launch_bounds__(256) void k_stem(
    const float* __restrict__ x, const float* __restrict__ w,
    const float* __restrict__ bg, const float* __restrict__ bb,
    const float* __restrict__ bm, const float* __restrict__ bv,
    float* __restrict__ out) {
  int i = blockIdx.x * blockDim.x + threadIdx.x;
  const int total = 8 * 32 * 112 * 112;
  if (i >= total) return;
  int ox = i % 112, oy = (i / 112) % 112;
  int co = (i / (112 * 112)) % 32, b = i / (112 * 112 * 32);
  float acc = 0.0f;
#pragma unroll
  for (int ci = 0; ci < 3; ++ci) {
    const float* xp = x + ((size_t)b * 3 + ci) * 224 * 224;
    const float* wp = w + (co * 3 + ci) * 9;
#pragma unroll
    for (int fy = 0; fy < 3; ++fy) {
      int iy = oy * 2 - 1 + fy;
      if ((unsigned)iy >= 224u) continue;
#pragma unroll
      for (int fx = 0; fx < 3; ++fx) {
        int ix = ox * 2 - 1 + fx;
        if ((unsigned)ix >= 224u) continue;
        acc += xp[iy * 224 + ix] * wp[fy * 3 + fx];
      }
    }
  }
  float s = bg[co] * rsqrtf(bv[co] + BN_EPS);
  float t = bb[co] - bm[co] * s;
  out[i] = fmaxf(acc * s + t, 0.0f);
}

// depthwise conv3x3 (stride 1 or 2) pad1, per-channel BN+ReLU
__global__ __launch_bounds__(256) void k_dw(
    const float* __restrict__ in, const float* __restrict__ w,
    const float* __restrict__ bg, const float* __restrict__ bb,
    const float* __restrict__ bm, const float* __restrict__ bv,
    float* __restrict__ out, int C, int Hin, int Win, int stride, int Ho, int Wo) {
  long i = blockIdx.x * (long)blockDim.x + threadIdx.x;
  long total = (long)8 * C * Ho * Wo;
  if (i >= total) return;
  int ox = (int)(i % Wo);
  int oy = (int)((i / Wo) % Ho);
  int c  = (int)((i / ((long)Wo * Ho)) % C);
  int b  = (int)(i / ((long)Wo * Ho * C));
  const float* ip = in + ((size_t)b * C + c) * Hin * Win;
  const float* wp = w + c * 9;
  float acc = 0.0f;
#pragma unroll
  for (int fy = 0; fy < 3; ++fy) {
    int iy = oy * stride - 1 + fy;
    if ((unsigned)iy >= (unsigned)Hin) continue;
#pragma unroll
    for (int fx = 0; fx < 3; ++fx) {
      int ix = ox * stride - 1 + fx;
      if ((unsigned)ix >= (unsigned)Win) continue;
      acc += ip[iy * Win + ix] * wp[fy * 3 + fx];
    }
  }
  float s = bg[c] * rsqrtf(bv[c] + BN_EPS);
  float t = bb[c] - bm[c] * s;
  out[i] = fmaxf(acc * s + t, 0.0f);
}

// pointwise 1x1 Ci->Co, BN+ReLU (backbone sizes are small; simple fp32)
__global__ __launch_bounds__(256) void k_pw(
    const float* __restrict__ in, const float* __restrict__ w,
    const float* __restrict__ bg, const float* __restrict__ bb,
    const float* __restrict__ bm, const float* __restrict__ bv,
    float* __restrict__ out, int Ci, int Co, int HW) {
  long i = blockIdx.x * (long)blockDim.x + threadIdx.x;
  long total = (long)8 * Co * HW;
  if (i >= total) return;
  int hw = (int)(i % HW);
  int co = (int)((i / HW) % Co);
  int b  = (int)(i / ((long)HW * Co));
  const float* ip = in + (size_t)b * Ci * HW + hw;
  const float* wp = w + (size_t)co * Ci;
  float acc = 0.0f;
  for (int ci = 0; ci < Ci; ++ci) acc += ip[(size_t)ci * HW] * wp[ci];
  float s = bg[co] * rsqrtf(bv[co] + BN_EPS);
  float t = bb[co] - bm[co] * s;
  out[i] = fmaxf(acc * s + t, 0.0f);
}

// ------------------------- head ---------------------------------------------

// Per-head depthwise 3x3 s2 pad1 over h[8,512,14,14] -> bf16 d[k][m=b*49+hw][c]
// (M-major, K-contiguous layout so WMMA A-fragments are contiguous 16B loads)
__global__ __launch_bounds__(256) void k_head_dw(
    const float* __restrict__ h, const float* __restrict__ w,
    const float* __restrict__ bnst, unsigned short* __restrict__ dbf) {
  long i = blockIdx.x * (long)blockDim.x + threadIdx.x;
  const long total = (long)256 * 8 * 49 * 512;
  if (i >= total) return;
  int c = (int)(i & 511);
  long r = i >> 9;
  int ohw = (int)(r % 49); r /= 49;
  int b = (int)(r & 7);
  int k = (int)(r >> 3);
  int oy = ohw / 7, ox = ohw % 7;
  const float* ip = h + ((size_t)b * 512 + c) * 196;
  const float* wp = w + ((size_t)k * 512 + c) * 9;
  float acc = 0.0f;
#pragma unroll
  for (int fy = 0; fy < 3; ++fy) {
    int iy = oy * 2 - 1 + fy;
    if ((unsigned)iy >= 14u) continue;
#pragma unroll
    for (int fx = 0; fx < 3; ++fx) {
      int ix = ox * 2 - 1 + fx;
      if ((unsigned)ix >= 14u) continue;
      acc += ip[iy * 14 + ix] * wp[fy * 3 + fx];
    }
  }
  float s = bnst[((size_t)k * 512 + c) * 2];
  float t = bnst[((size_t)k * 512 + c) * 2 + 1];
  dbf[i] = f2bf(fmaxf(acc * s + t, 0.0f));
}

// Per-head GEMM: d[k][392x512]bf16 x pw_w[k][1024x512]^T, BN+ReLU, 49-way
// spatial reduction per batch into g[k][8][1024].
// Workgroup = 4 waves: one 16(M)x64(N) strip per wave, 4 M-tiles per block.
// The shared 64x512 bf16 B-strip (64KB) is staged once per block into LDS
// with GLOBAL_LOAD_ASYNC_TO_LDS_B128; fragments then come from ds_load_b128.
__global__ __launch_bounds__(128) void k_head_gemm(
    const unsigned short* __restrict__ dbf, const unsigned short* __restrict__ wbf,
    const float* __restrict__ bnst, float* __restrict__ g) {
  __shared__ __align__(64) unsigned short lds_b[64 * 512];   // 64 KB

  const int ng  = blockIdx.x;        // 0..15  (N group of 64)
  const int mg  = blockIdx.y;        // 0..6   (group of 4 M-tiles)
  const int k   = blockIdx.z;        // 0..255 (head)
  const int tid = threadIdx.x;       // 0..127
  const int l   = tid & 31;
  const int wv  = tid >> 5;
  const int lo  = l & 15, hi = l >> 4;

  // ---- cooperative async copy: B strip [64 x 512] bf16 -> LDS -------------
  {
    const unsigned short* bsrc = wbf + ((size_t)k * 1024 + ng * 64) * 512;
#pragma unroll 4
    for (int i = 0; i < 32; ++i) {
      int c = tid + (i << 7);                               // 16B chunk id
      unsigned loff = (unsigned)(size_t)(&lds_b[c * 8]);    // LDS byte addr
      async_copy_b128(loff, (unsigned long long)(bsrc + c * 8));
    }
    wait_asynccnt0();
  }
  __syncthreads();

  const int mt = mg * 4 + wv;       // wave-uniform M-tile id
  if (mt < 25) {
    // A fragment addressing (16-bit A 16x32 layout): lane row m=lo,
    // halves at K = hi*8 + {0..7} and hi*8 + {16..23} -> two 16B loads.
    int m_a = mt * 16 + lo;
    if (m_a > 391) m_a = 391;       // clamp; masked in epilogue
    const unsigned short* abase = dbf + ((size_t)k * 392 + m_a) * 512 + hi * 8;

    AccF acc[4] = {};
    for (int kk = 0; kk < 512; kk += 32) {
      // WGP-scope prefetch of the A stream (locality 3 -> fill near caches)
      __builtin_prefetch(abase + kk + 64, 0, 3);            // global_prefetch_b8
      FragBF a;
      a.q[0] = *(const u32x4*)(abase + kk);
      a.q[1] = *(const u32x4*)(abase + kk + 16);
#pragma unroll
      for (int t = 0; t < 4; ++t) {
        // B fragment (16-bit B 32x16 layout): lane col n=lo, K=hi*8+{0..7}, hi*8+{16..23}
        const unsigned short* lp = &lds_b[(t * 16 + lo) * 512 + kk + hi * 8];
        FragBF bfr;
        bfr.q[0] = *(const u32x4*)(lp);                     // ds_load_b128
        bfr.q[1] = *(const u32x4*)(lp + 16);
        acc[t].v = __builtin_amdgcn_wmma_f32_16x16x32_bf16(
            false, a.v, false, bfr.v, (short)0, acc[t].v, false, false);
      }
    }

    // Epilogue: D layout lane col n=l%16, VGPR r -> row m = r + 8*(l>>4).
    // BN+ReLU, in-register reduction of same-batch rows, one atomic/segment.
#pragma unroll
    for (int t = 0; t < 4; ++t) {
      int n = ng * 64 + t * 16 + lo;
      float s  = bnst[((size_t)k * 1024 + n) * 2];
      float tt = bnst[((size_t)k * 1024 + n) * 2 + 1];
      float cur = 0.0f;
      int curb = -1;
#pragma unroll
      for (int r = 0; r < 8; ++r) {
        int m = mt * 16 + hi * 8 + r;
        if (m >= 392) break;
        int b = m / 49;
        float val = fmaxf(acc[t].f[r] * s + tt, 0.0f);
        if (b != curb) {
          if (curb >= 0) atomicAdd(&g[((size_t)k * 8 + curb) * 1024 + n], cur);
          curb = b; cur = 0.0f;
        }
        cur += val;
      }
      if (curb >= 0) atomicAdd(&g[((size_t)k * 8 + curb) * 1024 + n], cur);
    }
  }
}

// Final FC: out[k,b,o] = (g[k,b,:]/49) . fc_w[k,o,:] + fc_b[k,o]
__global__ __launch_bounds__(256) void k_fc(
    const float* __restrict__ g, const float* __restrict__ fw,
    const float* __restrict__ fb, float* __restrict__ out) {
  int i = blockIdx.x * blockDim.x + threadIdx.x;
  if (i >= 256 * 8 * 2) return;
  int o = i & 1, b = (i >> 1) & 7, k = i >> 4;
  const float* gv = g + ((size_t)k * 8 + b) * 1024;
  const float* wv = fw + ((size_t)k * 2 + o) * 1024;
  float acc = 0.0f;
  for (int d = 0; d < 1024; ++d) acc += gv[d] * wv[d];
  out[i] = acc * (1.0f / 49.0f) + fb[k * 2 + o];
}

// ------------------------- host-side launch --------------------------------

static inline long cdivl(long a, long b) { return (a + b - 1) / b; }

extern "C" void kernel_launch(void* const* d_in, const int* in_sizes, int n_in,
                              void* d_out, int out_size, void* d_ws, size_t ws_size,
                              hipStream_t stream) {
  (void)in_sizes; (void)n_in; (void)out_size; (void)ws_size;

  // --- input unpack (setup_inputs flatten order) ---
  const float* x      = (const float*)d_in[0];
  const float* stem_w = (const float*)d_in[1];
  const float* stem_bn[4] = {(const float*)d_in[2], (const float*)d_in[3],
                             (const float*)d_in[4], (const float*)d_in[5]};
  // blocks: base = 6 + 10*j : dw_w, dw_bn(4), pw_w, pw_bn(4)
  const float* hdw_w  = (const float*)d_in[66];
  const float* hdw_bn[4] = {(const float*)d_in[67], (const float*)d_in[68],
                            (const float*)d_in[69], (const float*)d_in[70]};
  const float* hpw_w  = (const float*)d_in[71];
  const float* hpw_bn[4] = {(const float*)d_in[72], (const float*)d_in[73],
                            (const float*)d_in[74], (const float*)d_in[75]};
  const float* fc_w   = (const float*)d_in[76];
  const float* fc_b   = (const float*)d_in[77];
  float* out = (float*)d_out;

  // --- workspace carve-up ---
  char* ws = (char*)d_ws;
  size_t off = 0;
  auto carve = [&](size_t bytes) {
    void* p = ws + off;
    off += (bytes + 255) & ~(size_t)255;
    return p;
  };
  float* bufA = (float*)carve((size_t)8 * 64 * 112 * 112 * 4);   // pw outputs
  float* bufB = (float*)carve((size_t)8 * 128 * 56 * 56 * 4);    // dw outputs
  unsigned short* dbf = (unsigned short*)carve((size_t)256 * 392 * 512 * 2);
  unsigned short* wbf = (unsigned short*)carve((size_t)256 * 1024 * 512 * 2);
  float* gbuf = (float*)carve((size_t)256 * 8 * 1024 * 4);
  float* pwst = (float*)carve((size_t)256 * 1024 * 2 * 4);
  float* dwst = (float*)carve((size_t)256 * 512 * 2 * 4);

  const int TB = 256;

  // --- prep: BN folds, bf16 weight cast, zero pool accumulator ---
  {
    int n = 256 * 1024;
    k_bn_fold<<<cdivl(n, TB), TB, 0, stream>>>(hpw_bn[0], hpw_bn[1], hpw_bn[2], hpw_bn[3], pwst, n);
    n = 256 * 512;
    k_bn_fold<<<cdivl(n, TB), TB, 0, stream>>>(hdw_bn[0], hdw_bn[1], hdw_bn[2], hdw_bn[3], dwst, n);
    long n4 = (long)256 * 1024 * 512 / 4;
    k_cast_bf16x4<<<cdivl(n4, TB), TB, 0, stream>>>(hpw_w, wbf, n4);
    long nz = (long)256 * 8 * 1024;
    k_zero<<<cdivl(nz, TB), TB, 0, stream>>>(gbuf, nz);
  }

  // --- backbone ---
  k_stem<<<cdivl((long)8 * 32 * 112 * 112, TB), TB, 0, stream>>>(
      x, stem_w, stem_bn[0], stem_bn[1], stem_bn[2], stem_bn[3], bufA);

  struct Blk { int ci, co, hin, stride; };
  const Blk blks[6] = {
      {32, 64, 112, 1}, {64, 128, 112, 2}, {128, 128, 56, 1},
      {128, 256, 56, 2}, {256, 512, 28, 2}, {512, 512, 14, 1}};
  for (int j = 0; j < 6; ++j) {
    const int base = 6 + 10 * j;
    const float* dw_w = (const float*)d_in[base + 0];
    const float* dbn0 = (const float*)d_in[base + 1];
    const float* dbn1 = (const float*)d_in[base + 2];
    const float* dbn2 = (const float*)d_in[base + 3];
    const float* dbn3 = (const float*)d_in[base + 4];
    const float* pw_w = (const float*)d_in[base + 5];
    const float* pbn0 = (const float*)d_in[base + 6];
    const float* pbn1 = (const float*)d_in[base + 7];
    const float* pbn2 = (const float*)d_in[base + 8];
    const float* pbn3 = (const float*)d_in[base + 9];
    const int hin = blks[j].hin, s = blks[j].stride;
    const int ho = (s == 1) ? hin : hin / 2;
    const int hw = ho * ho;
    long ndw = (long)8 * blks[j].ci * hw;
    k_dw<<<cdivl(ndw, TB), TB, 0, stream>>>(bufA, dw_w, dbn0, dbn1, dbn2, dbn3,
                                            bufB, blks[j].ci, hin, hin, s, ho, ho);
    long npw = (long)8 * blks[j].co * hw;
    k_pw<<<cdivl(npw, TB), TB, 0, stream>>>(bufB, pw_w, pbn0, pbn1, pbn2, pbn3,
                                            bufA, blks[j].ci, blks[j].co, hw);
  }
  // bufA now holds h [8,512,14,14]

  // --- heads ---
  k_head_dw<<<cdivl((long)256 * 8 * 49 * 512, TB), TB, 0, stream>>>(bufA, hdw_w, dwst, dbf);

  dim3 gg(16, 7, 256);  // Ngroups(64) x Mtile-groups(4 waves) x heads
  k_head_gemm<<<gg, 128, 0, stream>>>(dbf, wbf, pwst, gbuf);

  k_fc<<<cdivl(256 * 8 * 2, TB), TB, 0, stream>>>(gbuf, fc_w, fc_b, out);
}
